// WNGat_54039278518420
// MI455X (gfx1250) — matrix operand order repaired and checked by reading, not compiled, not run on hardware
//
#include <hip/hip_runtime.h>
#include <hip/hip_bf16.h>
#include <math.h>

// ---------------------------------------------------------------------------
// 3-layer GAT for MI455X (gfx1250, wave32).
//   h = X@W (bf16 WMMA, f32 acc) -> per-node attn scores -> 3 edge passes
//   (atomic segment max, exp+atomic segment sum, normalize+atomic aggregate)
//   -> bias+ELU. Node state (~25.6MB/layer) is L2-resident (192MB L2), so the
//   edge gather/scatter runs at L2 speed; the GEMM stages X and W into LDS in
//   WMMA-fragment-swizzled bf16 layout so the inner loop is pure
//   ds_load_b128 + v_wmma_f32_16x16x32_bf16, and D tiles are transposed
//   through per-wave LDS so stores go out as coalesced global_store_b128.
// ---------------------------------------------------------------------------

typedef __attribute__((ext_vector_type(16))) __bf16 v16bf;
typedef __attribute__((ext_vector_type(8)))  float  v8f;

#define FDIM 128          // all three layers are 128 -> 128 (H=2, C=64)
#define NHEAD 2

__device__ __forceinline__ void atomicMaxF32(float* addr, float val) {
  // Sign-split trick: works for mixed-sign values when initialized to -inf.
  if (val >= 0.0f) {
    atomicMax((int*)addr, __float_as_int(val));
  } else {
    atomicMin((unsigned int*)addr, __float_as_uint(val));
  }
}

__device__ __forceinline__ unsigned pack_bf16(float lo, float hi) {
  union { __hip_bfloat162 h2; unsigned u; } cv;
  cv.h2 = __hip_bfloat162(__hip_bfloat16(lo), __hip_bfloat16(hi));  // cvt_pk
  return cv.u;
}

// ------------------------- h = X @ W (WMMA, bf16) ---------------------------
// Block: 256 threads = 8 waves; block computes a 128x128 output tile.
// LDS layout:
//   [0, 32KB)  A region: frag f = (mt*4+kt)*32 + lane, 16 bf16 (A 16x32 map)
//   [32,64KB)  B region: frag f = (nt*4+kt)*32 + lane, 16 bf16 (B 32x16 map)
//   [64,72KB)  per-wave 16x16 f32 store-transpose tiles (1KB per wave)
__global__ void __launch_bounds__(256)
gat_gemm_wmma(const float* __restrict__ X, const float* __restrict__ W,
              float* __restrict__ H, int N) {
  extern __shared__ char smem[];
  unsigned* Alds = (unsigned*)smem;             // 8192 dwords = 32 KB
  unsigned* Blds = Alds + 8192;                 // 8192 dwords = 32 KB
  float*    Tlds = (float*)(Alds + 16384);      // 8 x 256 f32 = 8 KB

  const int tid = threadIdx.x;
  const int r0  = blockIdx.x * 128;

  // ---- Stage A (X tile) swizzled: t indexes a bf16 pair --------------------
  for (int t = tid; t < 8192; t += 256) {
    const int p    = t & 7;          // pair within fragment (elements 2p,2p+1)
    const int f    = t >> 3;         // fragment id
    const int lane = f & 31;
    const int kt   = (f >> 5) & 3;
    const int mt   = f >> 7;
    const int lm   = lane & 15, kg = lane >> 4;
    const int j0   = 2 * p;
    const int k    = kt * 32 + ((j0 < 8) ? (kg * 8 + j0) : (16 + kg * 8 + (j0 - 8)));
    const int row  = r0 + mt * 16 + lm;
    float lo = 0.0f, hi = 0.0f;
    if (row < N) {
      const float2 v = *(const float2*)(X + (size_t)row * FDIM + k);
      lo = v.x; hi = v.y;
    }
    Alds[t] = pack_bf16(lo, hi);
  }
  // ---- Stage B (W) swizzled ------------------------------------------------
  for (int t = tid; t < 8192; t += 256) {
    const int p    = t & 7;
    const int f    = t >> 3;
    const int lane = f & 31;
    const int kt   = (f >> 5) & 3;
    const int nt   = f >> 7;
    const int lm   = lane & 15, kg = lane >> 4;
    const int n    = nt * 16 + lm;
    const int k    = kt * 32 + ((2 * p < 8) ? (kg * 8 + 2 * p) : (16 + kg * 8 + (2 * p - 8)));
    Blds[t] = pack_bf16(W[(size_t)k * FDIM + n], W[(size_t)(k + 1) * FDIM + n]);
  }
  __syncthreads();

  const int wave = tid >> 5;        // == mt (row tile of this wave)
  const int lane = tid & 31;
  const int lm   = lane & 15;
  const int kg   = lane >> 4;
  const int m0   = wave * 16;
  const bool fullTile = (r0 + 128) <= N;   // block-uniform

  // Hoist all 4 A fragments (K = 128) for this wave: 2x ds_load_b128 each.
  const v16bf* Afr = (const v16bf*)Alds;
  const v16bf* Bfr = (const v16bf*)Blds;
  v16bf afrag[4];
#pragma unroll
  for (int kt = 0; kt < 4; ++kt)
    afrag[kt] = Afr[(wave * 4 + kt) * 32 + lane];

  float* Tw = Tlds + wave * 256;    // this wave's private 16x16 f32 tile

#pragma unroll
  for (int nt = 0; nt < 8; ++nt) {
    v8f acc = {};
#pragma unroll
    for (int kt = 0; kt < 4; ++kt) {
      const v16bf bfrag = Bfr[(nt * 4 + kt) * 32 + lane];
      acc = __builtin_amdgcn_wmma_f32_16x16x32_bf16(
          false, afrag[kt], false, bfrag, (short)0, acc, false, false);
    }
    // D layout: VGPR v -> row m0 + kg*8 + v, col = nt*16 + lm.
    if (fullTile) {
      // Transpose 16x16 tile through per-wave LDS (same-wave DS ops are
      // in-order, no barrier needed), then coalesced b128 row stores.
#pragma unroll
      for (int v = 0; v < 8; ++v)
        Tw[(kg * 8 + v) * 16 + lm] = acc[v];
      const int rl = lane >> 1;            // 0..15: local row
      const int c0 = (lane & 1) * 8;       // 0 or 8: starting col
      const float4 s0 = *(const float4*)(Tw + rl * 16 + c0);
      const float4 s1 = *(const float4*)(Tw + rl * 16 + c0 + 4);
      float* dst = H + (size_t)(r0 + m0 + rl) * FDIM + nt * 16 + c0;
      ((float4*)dst)[0] = s0;
      ((float4*)dst)[1] = s1;
    } else {
      const int n = nt * 16 + lm;
#pragma unroll
      for (int v = 0; v < 8; ++v) {
        const int row = r0 + m0 + kg * 8 + v;
        if (row < N) H[(size_t)row * FDIM + n] = acc[v];
      }
    }
  }
}

// -------------------- per-node attention scores (e_src/e_dst) ---------------
// One wave per node: lanes 0-15 -> head 0, lanes 16-31 -> head 1; each lane
// takes 4 channels; shuffle-reduce within each 16-lane half.
__global__ void __launch_bounds__(256)
gat_node_scores(const float* __restrict__ H, const float* __restrict__ a_src,
                const float* __restrict__ a_dst, float* __restrict__ esrc,
                float* __restrict__ edst, int N) {
  const int wave = threadIdx.x >> 5;
  const int lane = threadIdx.x & 31;
  const int node = blockIdx.x * 8 + wave;
  if (node >= N) return;
  const int head = lane >> 4;
  const int q    = lane & 15;

  const float4 hv = ((const float4*)(H + (size_t)node * FDIM))[head * 16 + q];
  const float4 av = ((const float4*)a_src)[head * 16 + q];
  const float4 dv = ((const float4*)a_dst)[head * 16 + q];
  float ps = hv.x * av.x + hv.y * av.y + hv.z * av.z + hv.w * av.w;
  float pd = hv.x * dv.x + hv.y * dv.y + hv.z * dv.z + hv.w * dv.w;
#pragma unroll
  for (int m = 8; m >= 1; m >>= 1) {
    ps += __shfl_xor(ps, m, 32);
    pd += __shfl_xor(pd, m, 32);
  }
  if (q == 0) {
    esrc[node * NHEAD + head] = ps;
    edst[node * NHEAD + head] = pd;
  }
}

// ----------------------------- per-layer init -------------------------------
__global__ void gat_init(float* __restrict__ segmax, float* __restrict__ denom,
                         float* __restrict__ acc, int n2, int nacc) {
  const int stride = gridDim.x * blockDim.x;
  for (int j = blockIdx.x * blockDim.x + threadIdx.x; j < nacc; j += stride)
    acc[j] = 0.0f;
  for (int j = blockIdx.x * blockDim.x + threadIdx.x; j < n2; j += stride) {
    segmax[j] = __int_as_float(0xff800000u);  // -inf
    denom[j]  = 0.0f;
  }
}

// --------------------- edge pass 1: logits + segment max --------------------
__global__ void __launch_bounds__(256)
gat_edge_logit_max(const long long* __restrict__ ei,
                   const float* __restrict__ esrc, const float* __restrict__ edst,
                   float* __restrict__ logit, float* __restrict__ segmax,
                   int E, int Etot) {
  const int e = blockIdx.x * blockDim.x + threadIdx.x;
  if (e >= Etot) return;
  int s, d;
  if (e < E) { s = (int)ei[e]; d = (int)ei[(size_t)E + e]; }
  else       { s = d = e - E; }                 // self loop
#pragma unroll
  for (int h = 0; h < NHEAD; ++h) {
    float l = esrc[s * NHEAD + h] + edst[d * NHEAD + h];
    l = (l > 0.0f) ? l : 0.2f * l;              // leaky-relu, slope 0.2
    logit[(size_t)e * NHEAD + h] = l;
    atomicMaxF32(&segmax[d * NHEAD + h], l);
  }
}

// --------------------- edge pass 2: exp + segment sum -----------------------
__global__ void __launch_bounds__(256)
gat_edge_expsum(const long long* __restrict__ ei, float* __restrict__ logit,
                const float* __restrict__ segmax, float* __restrict__ denom,
                int E, int Etot) {
  const int e = blockIdx.x * blockDim.x + threadIdx.x;
  if (e >= Etot) return;
  const int d = (e < E) ? (int)ei[(size_t)E + e] : (e - E);
#pragma unroll
  for (int h = 0; h < NHEAD; ++h) {
    const float v = __expf(logit[(size_t)e * NHEAD + h] - segmax[d * NHEAD + h]);
    logit[(size_t)e * NHEAD + h] = v;           // in-place: logit -> expv
    atomicAdd(&denom[d * NHEAD + h], v);
  }
}

// --------------- edge pass 3: alpha * h[src] scatter-add --------------------
// One wave per edge; lane -> 4 consecutive channels (float4 gather, 4 atomics).
__global__ void __launch_bounds__(256)
gat_edge_aggregate(const long long* __restrict__ ei, const float* __restrict__ H,
                   const float* __restrict__ expv, const float* __restrict__ denom,
                   float* __restrict__ acc, int E, long long Etot) {
  const int wave = threadIdx.x >> 5;
  const int lane = threadIdx.x & 31;
  const long long e = (long long)blockIdx.x * 8 + wave;
  if (e >= Etot) return;
  int s, d;
  if (e < E) { s = (int)ei[e]; d = (int)ei[(size_t)E + e]; }
  else       { s = d = (int)(e - E); }
  const int head = lane >> 4;                   // channels 0-63 / 64-127
  const float alpha = expv[e * NHEAD + head] / denom[d * NHEAD + head];

  const float* hrow = H + (size_t)s * FDIM;
  __builtin_prefetch(hrow, 0, 1);               // -> global_prefetch_b8
  const float4 hv = ((const float4*)hrow)[lane];
  float* o = acc + (size_t)d * FDIM + lane * 4;
  atomicAdd(o + 0, hv.x * alpha);
  atomicAdd(o + 1, hv.y * alpha);
  atomicAdd(o + 2, hv.z * alpha);
  atomicAdd(o + 3, hv.w * alpha);
}

// ----------------------------- bias + ELU -----------------------------------
__global__ void __launch_bounds__(256)
gat_bias_elu(float* __restrict__ acc, const float* __restrict__ b, int total) {
  const int i = blockIdx.x * blockDim.x + threadIdx.x;
  if (i >= total) return;
  const float v = acc[i] + b[i & (FDIM - 1)];
  acc[i] = (v > 0.0f) ? v : (__expf(v) - 1.0f);
}

// ---------------------------------------------------------------------------
extern "C" void kernel_launch(void* const* d_in, const int* in_sizes, int n_in,
                              void* d_out, int out_size, void* d_ws, size_t ws_size,
                              hipStream_t stream) {
  const float*     x  = (const float*)d_in[0];
  const long long* ei = (const long long*)d_in[1];
  const float* Wl[3] = {(const float*)d_in[2], (const float*)d_in[6],  (const float*)d_in[10]};
  const float* As[3] = {(const float*)d_in[3], (const float*)d_in[7],  (const float*)d_in[11]};
  const float* Ad[3] = {(const float*)d_in[4], (const float*)d_in[8],  (const float*)d_in[12]};
  const float* Bl[3] = {(const float*)d_in[5], (const float*)d_in[9],  (const float*)d_in[13]};

  const int N    = in_sizes[0] / FDIM;    // 50000
  const int E    = in_sizes[1] / 2;       // 800000
  const int Etot = E + N;                 // + self loops

  // Workspace carve-up (floats): h, ping, pong, 4 small node arrays, logits.
  float* p = (float*)d_ws;
  float* h      = p; p += (size_t)N * FDIM;
  float* ping   = p; p += (size_t)N * FDIM;
  float* pong   = p; p += (size_t)N * FDIM;
  float* esrc   = p; p += (size_t)N * NHEAD;
  float* edst   = p; p += (size_t)N * NHEAD;
  float* segmax = p; p += (size_t)N * NHEAD;
  float* denom  = p; p += (size_t)N * NHEAD;
  float* logit  = p; p += (size_t)Etot * NHEAD;
  if ((size_t)((char*)p - (char*)d_ws) > ws_size) return;  // insufficient ws

  const size_t ldsBytes = 65536 + 8192;   // A + B swizzled + store-transpose
  const int gemmBlocks  = (N + 127) / 128;
  const int nodeWaveBlk = (N + 7) / 8;
  const int edgeBlocks  = (Etot + 255) / 256;
  const int aggBlocks   = (Etot + 7) / 8;
  const int eluBlocks   = (N * FDIM + 255) / 256;

  const float* in = x;
  float* outs[3] = {ping, pong, (float*)d_out};

  for (int L = 0; L < 3; ++L) {
    float* out = outs[L];
    gat_gemm_wmma<<<gemmBlocks, 256, ldsBytes, stream>>>(in, Wl[L], h, N);
    gat_node_scores<<<nodeWaveBlk, 256, 0, stream>>>(h, As[L], Ad[L], esrc, edst, N);
    gat_init<<<2048, 256, 0, stream>>>(segmax, denom, out, N * NHEAD, N * FDIM);
    gat_edge_logit_max<<<edgeBlocks, 256, 0, stream>>>(ei, esrc, edst, logit, segmax, E, Etot);
    gat_edge_expsum<<<edgeBlocks, 256, 0, stream>>>(ei, logit, segmax, denom, E, Etot);
    gat_edge_aggregate<<<aggBlocks, 256, 0, stream>>>(ei, h, logit, denom, out, E, (long long)Etot);
    gat_bias_elu<<<eluBlocks, 256, 0, stream>>>(out, Bl[L], N * FDIM);
    in = out;
  }
}
